// SparseBKLayerWithMoE_62319975465272
// MI455X (gfx1250) — matrix operand order repaired and checked by reading, not compiled, not run on hardware
//
#include <hip/hip_runtime.h>
#include <hip/hip_bf16.h>
#include <cstdint>
#include <cstddef>

#define DIM    768
#define HID    3072
#define NEXP   8
#define NTOK   4096     // B*N
#define SEQ    2048
#define NBATCH 2
#define KKEEP  819      // int(N*0.4)
#define MTILE  32
#define MAX_TILES 136   // 4096/32 + 8 partial tiles

typedef _Float16 half16 __attribute__((ext_vector_type(16)));
typedef float    float8 __attribute__((ext_vector_type(8)));

union FragU { half16 h; uint4 q[2]; };

__device__ __forceinline__ half16 frag2(const _Float16* p0, const _Float16* p1) {
  FragU f;
  f.q[0] = *(const uint4*)p0;
  f.q[1] = *(const uint4*)p1;
  return f.h;
}

__device__ __forceinline__ float gelu_tanh(float x) {
  const float c = 0.7978845608028654f;      // sqrt(2/pi)
  float x3 = x * x * x;
  return 0.5f * x * (1.0f + tanhf(c * (x + 0.044715f * x3)));
}

// ---------------------------------------------------------------- transpose+cast
// src: [E][rows][cols] f32  ->  dst: [E][cols][rows] f16
__global__ __launch_bounds__(256) void k_transpose_cast(
    const float* __restrict__ src, _Float16* __restrict__ dst, int rows, int cols) {
  __shared__ float tile[32][33];
  int e  = blockIdx.z;
  int r0 = blockIdx.y * 32, c0 = blockIdx.x * 32;
  const float* s = src + (size_t)e * rows * cols;
  _Float16*    d = dst + (size_t)e * rows * cols;
  int tx = threadIdx.x, ty = threadIdx.y;
  for (int i = ty; i < 32; i += 8)
    tile[i][tx] = s[(size_t)(r0 + i) * cols + c0 + tx];
  __syncthreads();
  for (int i = ty; i < 32; i += 8)
    d[(size_t)(c0 + i) * rows + r0 + tx] = (_Float16)tile[tx][i];
}

// ---------------------------------------------------------------- routing (wave per token)
__global__ __launch_bounds__(256) void k_routing(
    const float* __restrict__ x, const float* __restrict__ Wg, const float* __restrict__ bg,
    const float* __restrict__ Wm, const float* __restrict__ bm,
    int* __restrict__ eid, float* __restrict__ topp, float* __restrict__ scores) {
  int wv   = (int)((blockIdx.x * blockDim.x + threadIdx.x) >> 5);
  int lane = threadIdx.x & 31;
  if (wv >= NTOK) return;
  const float* xr = x + (size_t)wv * DIM;
  float g[NEXP] = {0, 0, 0, 0, 0, 0, 0, 0};
  float sm = 0.0f;
  for (int d = lane; d < DIM; d += 32) {
    float xv = xr[d];
    const float* wg = Wg + d * NEXP;
#pragma unroll
    for (int e = 0; e < NEXP; ++e) g[e] += xv * wg[e];
    sm += xv * Wm[d];
  }
#pragma unroll
  for (int off = 16; off >= 1; off >>= 1) {
#pragma unroll
    for (int e = 0; e < NEXP; ++e) g[e] += __shfl_xor(g[e], off, 32);
    sm += __shfl_xor(sm, off, 32);
  }
  if (lane == 0) {
    float mx = -1e30f; int be = 0;
    float l[NEXP];
#pragma unroll
    for (int e = 0; e < NEXP; ++e) {
      l[e] = g[e] + bg[e];
      if (l[e] > mx) { mx = l[e]; be = e; }   // first max (jnp.argmax semantics)
    }
    float sum = 0.0f;
#pragma unroll
    for (int e = 0; e < NEXP; ++e) sum += expf(l[e] - mx);
    topp[wv]   = 1.0f / sum;                  // exp(mx-mx)/sum
    eid[wv]    = be;
    scores[wv] = 1.0f / (1.0f + expf(-(sm + bm[0])));
  }
}

// ---------------------------------------------------------------- compaction (1 block)
__global__ __launch_bounds__(256) void k_compact(
    const int* __restrict__ eid, int* __restrict__ perm,
    int4* __restrict__ tiles, int* __restrict__ meta) {
  __shared__ int cnt[NEXP], off[NEXP], cur[NEXP];
  int tid = threadIdx.x;
  if (tid < NEXP) cnt[tid] = 0;
  __syncthreads();
  for (int t = tid; t < NTOK; t += 256) atomicAdd(&cnt[eid[t]], 1);
  __syncthreads();
  if (tid == 0) {
    int o = 0, nt = 0;
    for (int e = 0; e < NEXP; ++e) { off[e] = o; cur[e] = o; o += cnt[e]; }
    for (int e = 0; e < NEXP; ++e)
      for (int s = 0; s < cnt[e]; s += MTILE) {
        int c = cnt[e] - s; if (c > MTILE) c = MTILE;
        tiles[nt] = make_int4(off[e] + s, c, e, 0);
        ++nt;
      }
    meta[0] = nt;
  }
  __syncthreads();
  for (int t = tid; t < NTOK; t += 256) {
    int pos = atomicAdd(&cur[eid[t]], 1);
    perm[pos] = t;                            // order within expert irrelevant: rows independent
  }
}

// ---------------------------------------------------------------- MoE FFN, WMMA f16
// 32-token tile per block; each wave owns a 16-wide N tile and both M-subtiles,
// so every B (weight) fragment is reused for two WMMAs.
__global__ __launch_bounds__(256) void k_ffn(
    const float* __restrict__ x,
    const _Float16* __restrict__ w1t,   // [E][H][D]
    const float* __restrict__ b1,       // [E][H]
    const _Float16* __restrict__ w2t,   // [E][D][H]
    const float* __restrict__ b2,       // [E][D]
    const int* __restrict__ perm, const float* __restrict__ topp,
    const int4* __restrict__ tiles, const int* __restrict__ meta,
    float* __restrict__ ffn_out) {
  __shared__ _Float16 Ax[MTILE][DIM + 8];    // 49.7 KB, row stride 388 dwords -> 4-bank skew
  __shared__ _Float16 Ht[MTILE][HID + 8];    // 197.1 KB
  __shared__ int   tok[MTILE];
  __shared__ float tsc[MTILE];

  int nt = meta[0];
  int bid = blockIdx.x;
  if (bid >= nt) return;
  int4 tl = tiles[bid];
  int start = tl.x, cntv = tl.y, e = tl.z;
  int tid = threadIdx.x;
  if (tid < MTILE) {
    int t = (tid < cntv) ? perm[start + tid] : -1;
    tok[tid] = t;
    tsc[tid] = (t >= 0) ? topp[t] : 0.0f;
  }
  __syncthreads();

  // gather X tile -> f16 LDS (coalesced along D)
  for (int i = tid; i < MTILE * DIM; i += 256) {
    int m = i / DIM, c = i - m * DIM;
    int t = tok[m];
    float xv = (t >= 0) ? x[(size_t)t * DIM + c] : 0.0f;
    Ax[m][c] = (_Float16)xv;
  }
  __syncthreads();

  int lane = tid & 31;
  int wave = tid >> 5;
  int n    = lane & 15;     // column within 16-wide N tile / row within M-subtile
  int sel  = lane >> 4;     // lane half select

  const _Float16* w1e = w1t + (size_t)e * HID * DIM;
  const float*    b1e = b1 + e * HID;

  // ---- layer 1: H = gelu(X @ W1 + b1) ----
  for (int ht = wave; ht < HID / 16; ht += 8) {
    int h0 = ht * 16;
    float bb = b1e[h0 + n];
    float8 acc0 = {bb, bb, bb, bb, bb, bb, bb, bb};
    float8 acc1 = acc0;
    const _Float16* arow0 = &Ax[n][0];
    const _Float16* arow1 = &Ax[16 + n][0];
    const _Float16* bcol  = w1e + (size_t)(h0 + n) * DIM + sel * 16;
    // warm next weight tile for this wave (global_prefetch)
    if (ht + 8 < HID / 16)
      __builtin_prefetch(w1e + (size_t)(h0 + 128 + n) * DIM + sel * 16, 0, 0);
#pragma unroll 4
    for (int k0 = 0; k0 < DIM; k0 += 32) {
      half16 b  = frag2(bcol + k0 - sel * 8,       bcol + k0 + 16 - sel * 8);
      half16 a0 = frag2(arow0 + k0 + sel * 8,      arow0 + k0 + 16 + sel * 8);
      half16 a1 = frag2(arow1 + k0 + sel * 8,      arow1 + k0 + 16 + sel * 8);
      acc0 = __builtin_amdgcn_wmma_f32_16x16x32_f16(false, a0, false, b, (short)0, acc0, false, false);
      acc1 = __builtin_amdgcn_wmma_f32_16x16x32_f16(false, a1, false, b, (short)0, acc1, false, false);
    }
#pragma unroll
    for (int r = 0; r < 8; ++r) {
      int m = r + sel * 8;
      Ht[m][h0 + n]      = (_Float16)gelu_tanh(acc0[r]);
      Ht[16 + m][h0 + n] = (_Float16)gelu_tanh(acc1[r]);
    }
  }
  __syncthreads();

  const _Float16* w2e = w2t + (size_t)e * DIM * HID;
  const float*    b2e = b2 + e * DIM;

  // ---- layer 2: Y = (H @ W2 + b2) * top_p ----
  for (int dt = wave; dt < DIM / 16; dt += 8) {
    int d0 = dt * 16;
    float bb = b2e[d0 + n];
    float8 acc0 = {bb, bb, bb, bb, bb, bb, bb, bb};
    float8 acc1 = acc0;
    const _Float16* arow0 = &Ht[n][0];
    const _Float16* arow1 = &Ht[16 + n][0];
    const _Float16* bcol  = w2e + (size_t)(d0 + n) * HID + sel * 16;
    if (dt + 8 < DIM / 16)
      __builtin_prefetch(w2e + (size_t)(d0 + 128 + n) * HID + sel * 16, 0, 0);
#pragma unroll 4
    for (int k0 = 0; k0 < HID; k0 += 32) {
      half16 b  = frag2(bcol + k0 - sel * 8,       bcol + k0 + 16 - sel * 8);
      half16 a0 = frag2(arow0 + k0 + sel * 8,      arow0 + k0 + 16 + sel * 8);
      half16 a1 = frag2(arow1 + k0 + sel * 8,      arow1 + k0 + 16 + sel * 8);
      acc0 = __builtin_amdgcn_wmma_f32_16x16x32_f16(false, a0, false, b, (short)0, acc0, false, false);
      acc1 = __builtin_amdgcn_wmma_f32_16x16x32_f16(false, a1, false, b, (short)0, acc1, false, false);
    }
#pragma unroll
    for (int r = 0; r < 8; ++r) {
      int m = r + sel * 8;
      int t0 = tok[m];
      int t1 = tok[16 + m];
      if (t0 >= 0) ffn_out[(size_t)t0 * DIM + d0 + n] = acc0[r] * tsc[m];
      if (t1 >= 0) ffn_out[(size_t)t1 * DIM + d0 + n] = acc1[r] * tsc[16 + m];
    }
  }
}

// ---------------------------------------------------------------- v = clip(ffn @ Wv + bv)
__global__ __launch_bounds__(256) void k_vpot(
    const float* __restrict__ ffn, const float* __restrict__ Wv,
    const float* __restrict__ bv, float* __restrict__ vout) {
  int wv   = (int)((blockIdx.x * blockDim.x + threadIdx.x) >> 5);
  int lane = threadIdx.x & 31;
  if (wv >= NTOK) return;
  const float* r = ffn + (size_t)wv * DIM;
  float s = 0.0f;
  for (int d = lane; d < DIM; d += 32) s += r[d] * Wv[d];
#pragma unroll
  for (int off = 16; off >= 1; off >>= 1) s += __shfl_xor(s, off, 32);
  if (lane == 0) vout[wv] = fminf(fmaxf(s + bv[0], -3.0f), 3.0f);
}

// ---------------------------------------------------------------- exact K-th largest via bitonic sort
__global__ __launch_bounds__(256) void k_thr(const float* __restrict__ scores,
                                             float* __restrict__ thr) {
  __shared__ float s[SEQ];
  int b = blockIdx.x, tid = threadIdx.x;
  for (int i = tid; i < SEQ; i += 256) s[i] = scores[b * SEQ + i];
  __syncthreads();
  for (int k = 2; k <= SEQ; k <<= 1)
    for (int j = k >> 1; j > 0; j >>= 1) {
      for (int i = tid; i < SEQ; i += 256) {
        int ixj = i ^ j;
        if (ixj > i) {
          bool up = ((i & k) == 0);
          float a = s[i], c = s[ixj];
          if ((a > c) == up) { s[i] = c; s[ixj] = a; }
        }
      }
      __syncthreads();
    }
  if (tid == 0) thr[b] = s[SEQ - KKEEP];   // ascending sort -> K-th largest
}

// ---------------------------------------------------------------- BK resolvent: chunked Moebius scan
struct CMat { float r00, i00, r01, i01, r10, i10, r11, i11; };

__device__ __forceinline__ CMat cident() { CMat m = {1, 0, 0, 0, 0, 0, 1, 0}; return m; }

__device__ __forceinline__ CMat cmul(const CMat& A, const CMat& B) {
  CMat R;
  R.r00 = A.r00*B.r00 - A.i00*B.i00 + A.r01*B.r10 - A.i01*B.i10;
  R.i00 = A.r00*B.i00 + A.i00*B.r00 + A.r01*B.i10 + A.i01*B.r10;
  R.r01 = A.r00*B.r01 - A.i00*B.i01 + A.r01*B.r11 - A.i01*B.i11;
  R.i01 = A.r00*B.i01 + A.i00*B.r01 + A.r01*B.i11 + A.i01*B.r11;
  R.r10 = A.r10*B.r00 - A.i10*B.i00 + A.r11*B.r10 - A.i11*B.i10;
  R.i10 = A.r10*B.i00 + A.i10*B.r00 + A.r11*B.i10 + A.i11*B.r10;
  R.r11 = A.r10*B.r01 - A.i10*B.i01 + A.r11*B.r11 - A.i11*B.i11;
  R.i11 = A.r10*B.i01 + A.i10*B.r01 + A.r11*B.i11 + A.i11*B.r11;
  return R;
}

__device__ __forceinline__ void cnorm(CMat& A) {
  float m = fmaxf(fmaxf(fmaxf(fabsf(A.r00), fabsf(A.i00)), fmaxf(fabsf(A.r01), fabsf(A.i01))),
                  fmaxf(fmaxf(fabsf(A.r10), fabsf(A.i10)), fmaxf(fabsf(A.r11), fabsf(A.i11))));
  float s = 1.0f / fmaxf(m, 1e-30f);
  A.r00 *= s; A.i00 *= s; A.r01 *= s; A.i01 *= s;
  A.r10 *= s; A.i10 *= s; A.r11 *= s; A.i11 *= s;
}

__device__ __forceinline__ CMat cshfl_up(const CMat& A, int d) {
  CMat R;
  R.r00 = __shfl_up(A.r00, d, 32); R.i00 = __shfl_up(A.i00, d, 32);
  R.r01 = __shfl_up(A.r01, d, 32); R.i01 = __shfl_up(A.i01, d, 32);
  R.r10 = __shfl_up(A.r10, d, 32); R.i10 = __shfl_up(A.i10, d, 32);
  R.r11 = __shfl_up(A.r11, d, 32); R.i11 = __shfl_up(A.i11, d, 32);
  return R;
}

__device__ void bk_pass(const float* __restrict__ vb, int lane, bool rev,
                        float* __restrict__ arp, float* __restrict__ aip,
                        const float* __restrict__ arin, const float* __restrict__ aiin,
                        float* __restrict__ grp, float* __restrict__ gip) {
  const int CH = SEQ / 32;                    // 64-element chunk per lane
  // 1) per-chunk composite of M_i = [[d_i,-1],[1,0]] (left products)
  CMat C = cident();
  for (int j = 0; j < CH; ++j) {
    int pos = lane * CH + j;
    int idx = rev ? (SEQ - 1 - pos) : pos;
    float dr = vb[idx], di = -1.0f;           // ETA = 1
    float n0r = dr * C.r00 - di * C.i00 - C.r10;
    float n0i = dr * C.i00 + di * C.r00 - C.i10;
    float n1r = dr * C.r01 - di * C.i01 - C.r11;
    float n1i = dr * C.i01 + di * C.r01 - C.i11;
    C.r10 = C.r00; C.i10 = C.i00; C.r11 = C.r01; C.i11 = C.i01;
    C.r00 = n0r;   C.i00 = n0i;   C.r01 = n1r;   C.i01 = n1i;
    if ((j & 15) == 15) cnorm(C);
  }
  // 2) inclusive scan of composites across the wave, then exclusive prefix
  for (int s = 1; s < 32; s <<= 1) {
    CMat O = cshfl_up(C, s);
    if (lane >= s) { C = cmul(C, O); cnorm(C); }
  }
  CMat P = cshfl_up(C, 1);
  if (lane == 0) P = cident();
  // entry state (p,q) = P * (1,0);  1/a_entry = q/p = q*conj(p)/|p|^2
  float den  = P.r00 * P.r00 + P.i00 * P.i00;
  float invr = (P.r10 * P.r00 + P.i10 * P.i00) / den;
  float invi = (P.i10 * P.r00 - P.r10 * P.i00) / den;
  // 3) serial replay of the chunk
  for (int j = 0; j < CH; ++j) {
    int pos = lane * CH + j;
    int idx = rev ? (SEQ - 1 - pos) : pos;
    float dr = vb[idx];
    float xr = dr - invr;
    float xi = -1.0f - invi;
    if (!rev) {
      arp[idx] = xr; aip[idx] = xi;
    } else {
      float cr = arin[idx] + xr - dr;
      float ci = aiin[idx] + xi + 1.0f;       // -di = +1
      float dd = cr * cr + ci * ci;
      grp[idx] =  cr / dd;
      gip[idx] = -ci / dd;
    }
    float dn = xr * xr + xi * xi;             // Im(a) <= -1 -> dn >= 1
    invr =  xr / dn;
    invi = -xi / dn;
  }
}

__global__ __launch_bounds__(64) void k_bk(const float* __restrict__ v,
                                           float* __restrict__ ar_, float* __restrict__ ai_,
                                           float* __restrict__ gr_, float* __restrict__ gi_) {
  int lane = threadIdx.x & 31;
  int b    = threadIdx.x >> 5;                // one wave per batch
  const float* vb = v + b * SEQ;
  int off = b * SEQ;
  bk_pass(vb, lane, false, ar_ + off, ai_ + off, nullptr, nullptr, nullptr, nullptr);
  __threadfence_block();
  __syncthreads();
  bk_pass(vb, lane, true, nullptr, nullptr, ar_ + off, ai_ + off, gr_ + off, gi_ + off);
}

// ---------------------------------------------------------------- fused epilogue
__global__ __launch_bounds__(256) void k_final(
    const float* __restrict__ ffn, const float* __restrict__ gr, const float* __restrict__ gi,
    const float* __restrict__ scores, const float* __restrict__ thr,
    const float* __restrict__ Wo, const float* __restrict__ bo,
    const float* __restrict__ bk_scale, float* __restrict__ out) {
  size_t i = (size_t)blockIdx.x * 256 + threadIdx.x;
  if (i >= (size_t)NTOK * DIM) return;
  int t = (int)(i / DIM);
  int d = (int)(i - (size_t)t * DIM);
  int b = t >> 11;                            // t / SEQ
  float mk = (scores[t] >= thr[b]) ? 1.0f : 0.0f;
  float f0 = fminf(fmaxf(gr[t] * mk, -10.0f), 10.0f);
  float f1 = fminf(fmaxf(gi[t] * mk, -10.0f), 10.0f);
  out[i] = ffn[i] + bk_scale[0] * (f0 * Wo[d] + f1 * Wo[DIM + d] + bo[d]);
}

// ---------------------------------------------------------------- host
extern "C" void kernel_launch(void* const* d_in, const int* in_sizes, int n_in,
                              void* d_out, int out_size, void* d_ws, size_t ws_size,
                              hipStream_t stream) {
  const float* x   = (const float*)d_in[0];
  const float* Wg  = (const float*)d_in[1];
  const float* bg  = (const float*)d_in[2];
  const float* W1  = (const float*)d_in[3];
  const float* b1  = (const float*)d_in[4];
  const float* W2  = (const float*)d_in[5];
  const float* b2  = (const float*)d_in[6];
  const float* Wv  = (const float*)d_in[7];
  const float* bv  = (const float*)d_in[8];
  const float* Wm  = (const float*)d_in[9];
  const float* bm  = (const float*)d_in[10];
  const float* Wo  = (const float*)d_in[11];
  const float* bo  = (const float*)d_in[12];
  const float* bks = (const float*)d_in[13];
  float* out = (float*)d_out;

  char* wsp = (char*)d_ws;
  auto alloc = [&](size_t bytes) -> void* {
    void* p = (void*)wsp;
    wsp += (bytes + 255) & ~(size_t)255;
    return p;
  };
  _Float16* w1t   = (_Float16*)alloc((size_t)NEXP * HID * DIM * 2);
  _Float16* w2t   = (_Float16*)alloc((size_t)NEXP * DIM * HID * 2);
  float*    ffn   = (float*)alloc((size_t)NTOK * DIM * 4);
  float*    topp  = (float*)alloc(NTOK * 4);
  int*      eid   = (int*)alloc(NTOK * 4);
  float*    scr   = (float*)alloc(NTOK * 4);
  int*      perm  = (int*)alloc(NTOK * 4);
  int4*     tiles = (int4*)alloc(MAX_TILES * sizeof(int4));
  int*      meta  = (int*)alloc(64 * 4);
  float*    vbuf  = (float*)alloc(NTOK * 4);
  float*    arw   = (float*)alloc(NTOK * 4);
  float*    aiw   = (float*)alloc(NTOK * 4);
  float*    grw   = (float*)alloc(NTOK * 4);
  float*    giw   = (float*)alloc(NTOK * 4);
  float*    thr   = (float*)alloc(64 * 4);

  dim3 tb(32, 8);
  // W1 (E,D,H) -> w1t (E,H,D);  W2 (E,H,D) -> w2t (E,D,H)
  k_transpose_cast<<<dim3(HID / 32, DIM / 32, NEXP), tb, 0, stream>>>(W1, w1t, DIM, HID);
  k_transpose_cast<<<dim3(DIM / 32, HID / 32, NEXP), tb, 0, stream>>>(W2, w2t, HID, DIM);

  k_routing<<<NTOK / 8, 256, 0, stream>>>(x, Wg, bg, Wm, bm, eid, topp, scr);
  k_compact<<<1, 256, 0, stream>>>(eid, perm, tiles, meta);
  k_ffn<<<MAX_TILES, 256, 0, stream>>>(x, w1t, b1, w2t, b2, perm, topp, tiles, meta, ffn);
  k_vpot<<<NTOK / 8, 256, 0, stream>>>(ffn, Wv, bv, vbuf);
  k_thr<<<NBATCH, 256, 0, stream>>>(scr, thr);
  k_bk<<<1, 64, 0, stream>>>(vbuf, arw, aiw, grw, giw);

  size_t total = (size_t)NTOK * DIM;
  k_final<<<(unsigned)((total + 255) / 256), 256, 0, stream>>>(
      ffn, grw, giw, scr, thr, Wo, bo, bks, out);
}
